// RTTBlock_30709016167137
// MI455X (gfx1250) — matrix operand order repaired and checked by reading, not compiled, not run on hardware
//
#include <hip/hip_runtime.h>
#include <hip/hip_bf16.h>

#define BB 2
#define NN 2048
#define DD 1024
#define HH 16
#define HDIM 64
#define EPSF 1e-10f

typedef __bf16 bf16;
typedef __attribute__((ext_vector_type(8)))  bf16  bf16x8;
typedef __attribute__((ext_vector_type(16))) bf16  v16bf;
typedef __attribute__((ext_vector_type(8)))  float v8f;

static __device__ inline v8f vzero8() {
  v8f v;
#pragma unroll
  for (int i = 0; i < 8; ++i) v[i] = 0.0f;
  return v;
}

static __device__ inline v16bf cat8(bf16x8 lo, bf16x8 hi) {
  return __builtin_shufflevector(lo, hi, 0,1,2,3,4,5,6,7,8,9,10,11,12,13,14,15);
}

// A operand (16x32 bf16). Lane (r,half): row r, K chunks {half*8..+7, 16+half*8..+7}
static __device__ inline v16bf ld_a(const bf16* base, int stride, int half, int r) {
  const bf16* p = base + r * stride + half * 8;
  bf16x8 lo = *(const bf16x8*)(p);
  bf16x8 hi = *(const bf16x8*)(p + 16);
  return cat8(lo, hi);
}

// B operand (32x16 bf16). Lane (r,half): column r, K chunks {half*8..+7, 16+half*8..+7}
static __device__ inline v16bf ld_b(const bf16* base, int stride, int half, int r) {
  const bf16* p = base + r * stride + half * 8;
  bf16x8 lo = *(const bf16x8*)(p);
  bf16x8 hi = *(const bf16x8*)(p + 16);
  return cat8(lo, hi);
}

static __device__ inline v8f wmma_bf16(v16bf a, v16bf b, v8f c) {
  return __builtin_amdgcn_wmma_f32_16x16x32_bf16(false, a, false, b, (short)0, c,
                                                 false, false);
}

// ---------------- fused QKV projection (bf16 WMMA, f32 accumulate) ----------------
// X (4096 x 1024) f32 ; W* (1024 x 1024) f32 ; out Q,K,V as (B,H,N,HD) bf16.
// Q pre-scaled by 1/sqrt(HD).
__global__ __launch_bounds__(256) void k_qkv(const float* __restrict__ X,
    const float* __restrict__ Wq, const float* __restrict__ bq,
    const float* __restrict__ Wk, const float* __restrict__ bk,
    const float* __restrict__ Wv, const float* __restrict__ bv,
    bf16* __restrict__ Q, bf16* __restrict__ K, bf16* __restrict__ V) {
  __shared__ alignas(16) bf16 Xs[64 * 32];
  __shared__ alignas(16) bf16 Ws[3][128 * 32];
  const int t = threadIdx.x;
  const int mb = blockIdx.x >> 3;
  const int nb = blockIdx.x & 7;
  const int mbase = mb * 64, nbase = nb * 128;
  const int w = t >> 5, lane = t & 31, r = lane & 15, half = lane >> 4;
  const int mt = w & 1, nt = w >> 1;            // wave tile: 32 rows x 32 cols
  const float* Wsrc[3] = {Wq, Wk, Wv};

  v8f acc[3][2][2];
#pragma unroll
  for (int q = 0; q < 3; ++q)
#pragma unroll
    for (int a = 0; a < 2; ++a)
#pragma unroll
      for (int bcol = 0; bcol < 2; ++bcol) acc[q][a][bcol] = vzero8();

  for (int kb = 0; kb < DD; kb += 32) {
    {  // stage X tile (64x32) f32 -> bf16
      int row = t >> 2, c0 = (t & 3) * 8;
      const float4* g = (const float4*)(X + (size_t)(mbase + row) * DD + kb + c0);
      float4 f0 = g[0], f1 = g[1];
      bf16* d = &Xs[row * 32 + c0];
      d[0]=(bf16)f0.x; d[1]=(bf16)f0.y; d[2]=(bf16)f0.z; d[3]=(bf16)f0.w;
      d[4]=(bf16)f1.x; d[5]=(bf16)f1.y; d[6]=(bf16)f1.z; d[7]=(bf16)f1.w;
    }
#pragma unroll
    for (int q = 0; q < 3; ++q) {  // stage W tile (128x32)
      int row = t >> 1, c0 = (t & 1) * 16;
      const float4* g = (const float4*)(Wsrc[q] + (size_t)(nbase + row) * DD + kb + c0);
      float4 f0 = g[0], f1 = g[1], f2 = g[2], f3 = g[3];
      bf16* d = &Ws[q][row * 32 + c0];
      d[0]=(bf16)f0.x;  d[1]=(bf16)f0.y;  d[2]=(bf16)f0.z;  d[3]=(bf16)f0.w;
      d[4]=(bf16)f1.x;  d[5]=(bf16)f1.y;  d[6]=(bf16)f1.z;  d[7]=(bf16)f1.w;
      d[8]=(bf16)f2.x;  d[9]=(bf16)f2.y;  d[10]=(bf16)f2.z; d[11]=(bf16)f2.w;
      d[12]=(bf16)f3.x; d[13]=(bf16)f3.y; d[14]=(bf16)f3.z; d[15]=(bf16)f3.w;
    }
    __syncthreads();
    v16bf a0 = ld_a(&Xs[(mt * 32 + 0) * 32], 32, half, r);
    v16bf a1 = ld_a(&Xs[(mt * 32 + 16) * 32], 32, half, r);
#pragma unroll
    for (int q = 0; q < 3; ++q) {
      v16bf b0 = ld_b(&Ws[q][(nt * 32 + 0) * 32], 32, half, r);
      v16bf b1 = ld_b(&Ws[q][(nt * 32 + 16) * 32], 32, half, r);
      acc[q][0][0] = wmma_bf16(a0, b0, acc[q][0][0]);
      acc[q][0][1] = wmma_bf16(a0, b1, acc[q][0][1]);
      acc[q][1][0] = wmma_bf16(a1, b0, acc[q][1][0]);
      acc[q][1][1] = wmma_bf16(a1, b1, acc[q][1][1]);
    }
    __syncthreads();
  }
  const float* bias[3] = {bq, bk, bv};
  bf16* outp[3] = {Q, K, V};
#pragma unroll
  for (int q = 0; q < 3; ++q) {
    const float scale = (q == 0) ? 0.125f : 1.0f;
#pragma unroll
    for (int ms = 0; ms < 2; ++ms)
#pragma unroll
      for (int ns = 0; ns < 2; ++ns) {
        int ng = nbase + nt * 32 + ns * 16 + r;
        float bb = bias[q][ng];
        int h = ng >> 6, hd = ng & 63;
#pragma unroll
        for (int v = 0; v < 8; ++v) {
          int mg = mbase + mt * 32 + ms * 16 + v + 8 * half;
          int bi = mg >> 11, tok = mg & (NN - 1);
          float val = (acc[q][ms][ns][v] + bb) * scale;
          outp[q][(((size_t)bi * HH + h) * NN + tok) * HDIM + hd] = (bf16)val;
        }
      }
  }
}

// ---------------- V transpose: (B,H,N,HD) -> (B,H,HD,N) bf16 ----------------
__global__ __launch_bounds__(256) void k_transV(const bf16* __restrict__ V,
                                                bf16* __restrict__ Vt) {
  __shared__ bf16 tile[64][72];
  const int blk = blockIdx.x;           // B*H*(N/64) = 1024
  const int bh = blk >> 5;
  const int n0 = (blk & 31) * 64;
  const int t = threadIdx.x;
  int row = t >> 2, c0 = (t & 3) * 16;
  const bf16* src = V + ((size_t)bh * NN + n0 + row) * HDIM + c0;
#pragma unroll
  for (int j = 0; j < 16; ++j) tile[row][c0 + j] = src[j];
  __syncthreads();
  int hd = t >> 2;
  bf16* dst = Vt + ((size_t)bh * HDIM + hd) * NN + n0 + c0;
#pragma unroll
  for (int j = 0; j < 16; ++j) dst[j] = tile[c0 + j][hd];
}

// ---------------- attention sweep 1: row max / sum-exp via WMMA scores ----------------
__global__ __launch_bounds__(256) void k_attn_stats(const bf16* __restrict__ Q,
    const bf16* __restrict__ K, float* __restrict__ Mst, float* __restrict__ Lst) {
  const int blk = blockIdx.x;           // B*H*(N/128) = 512
  const int bh = blk >> 4;
  const int qt = blk & 15;
  const int t = threadIdx.x, w = t >> 5, lane = t & 31, r = lane & 15, half = lane >> 4;
  const int q0 = qt * 128 + w * 16;
  const bf16* Qb = Q + ((size_t)bh * NN + q0) * HDIM;
  v16bf a0 = ld_a(Qb, HDIM, half, r);
  v16bf a1 = ld_a(Qb + 32, HDIM, half, r);
  float mrun[8], lrun[8];
#pragma unroll
  for (int v = 0; v < 8; ++v) { mrun[v] = -1.0e30f; lrun[v] = 0.0f; }
  const bf16* Kb = K + (size_t)bh * NN * HDIM;
  for (int kb = 0; kb < NN; kb += 16) {
    const bf16* kp = Kb + (size_t)kb * HDIM;
    v16bf b0 = ld_b(kp, HDIM, half, r);
    v16bf b1 = ld_b(kp + 32, HDIM, half, r);
    v8f s = vzero8();
    s = wmma_bf16(a0, b0, s);
    s = wmma_bf16(a1, b1, s);
#pragma unroll
    for (int v = 0; v < 8; ++v) {
      float val = s[v];
      float cm = val;
      cm = fmaxf(cm, __shfl_xor(cm, 1, 32));
      cm = fmaxf(cm, __shfl_xor(cm, 2, 32));
      cm = fmaxf(cm, __shfl_xor(cm, 4, 32));
      cm = fmaxf(cm, __shfl_xor(cm, 8, 32));
      float nm = fmaxf(mrun[v], cm);
      float e = __expf(val - nm);
      e += __shfl_xor(e, 1, 32);
      e += __shfl_xor(e, 2, 32);
      e += __shfl_xor(e, 4, 32);
      e += __shfl_xor(e, 8, 32);
      lrun[v] = lrun[v] * __expf(mrun[v] - nm) + e;
      mrun[v] = nm;
    }
  }
  if (r == 0) {
#pragma unroll
    for (int v = 0; v < 8; ++v) {
      int idx = bh * NN + q0 + v + 8 * half;
      Mst[idx] = mrun[v];
      Lst[idx] = lrun[v];
    }
  }
}

// ---------------- attention sweep 2: p = exp(s-m)/l ; out += p@V ; avg += p/H ----------------
__global__ __launch_bounds__(256) void k_attn_apply(const bf16* __restrict__ Q,
    const bf16* __restrict__ K, const bf16* __restrict__ Vt,
    const float* __restrict__ Mst, const float* __restrict__ Lst,
    float* __restrict__ avg, bf16* __restrict__ attnout) {
  __shared__ alignas(16) bf16 Ps[8][16][32];
  const int blk = blockIdx.x;           // 512
  const int bh = blk >> 4;
  const int qt = blk & 15;
  const int t = threadIdx.x, w = t >> 5, lane = t & 31, r = lane & 15, half = lane >> 4;
  const int q0 = qt * 128 + w * 16;
  const int b_ = bh / HH, h_ = bh % HH;
  const bf16* Qb = Q + ((size_t)bh * NN + q0) * HDIM;
  v16bf a0 = ld_a(Qb, HDIM, half, r);
  v16bf a1 = ld_a(Qb + 32, HDIM, half, r);
  float mrow[8], linv[8];
  size_t avgbase[8];
#pragma unroll
  for (int v = 0; v < 8; ++v) {
    int row = q0 + v + 8 * half;
    int idx = bh * NN + row;
    mrow[v] = Mst[idx];
    linv[v] = 1.0f / Lst[idx];
    avgbase[v] = ((size_t)b_ * NN + row) * NN;
  }
  v8f o[4];
#pragma unroll
  for (int n = 0; n < 4; ++n) o[n] = vzero8();
  const bf16* Kb = K + (size_t)bh * NN * HDIM;
  const bf16* Vb = Vt + (size_t)bh * HDIM * NN;
  for (int kb = 0; kb < NN; kb += 32) {
#pragma unroll
    for (int c = 0; c < 2; ++c) {
      int kc = kb + c * 16;
      const bf16* kp = Kb + (size_t)kc * HDIM;
      v16bf b0 = ld_b(kp, HDIM, half, r);
      v16bf b1 = ld_b(kp + 32, HDIM, half, r);
      v8f s = vzero8();
      s = wmma_bf16(a0, b0, s);
      s = wmma_bf16(a1, b1, s);
#pragma unroll
      for (int v = 0; v < 8; ++v) {
        float p = __expf(s[v] - mrow[v]) * linv[v];
        atomicAdd(&avg[avgbase[v] + kc + r], p * (1.0f / HH));
        Ps[w][v + 8 * half][c * 16 + r] = (bf16)p;
      }
    }
    asm volatile("s_wait_dscnt 0" ::: "memory");
    v16bf ap = ld_a(&Ps[w][0][0], 32, half, r);
#pragma unroll
    for (int n = 0; n < 4; ++n) {
      v16bf bv2 = ld_b(Vb + (size_t)(n * 16) * NN + kb, NN, half, r);
      o[n] = wmma_bf16(ap, bv2, o[n]);
    }
    asm volatile("s_wait_dscnt 0" ::: "memory");
  }
#pragma unroll
  for (int n = 0; n < 4; ++n)
#pragma unroll
    for (int v = 0; v < 8; ++v) {
      int tok = q0 + v + 8 * half;
      attnout[((size_t)b_ * NN + tok) * DD + h_ * HDIM + n * 16 + r] = (bf16)o[n][v];
    }
}

// ---------------- output projection: out = attnout @ Wo^T + bo (f32 result) ----------------
__global__ __launch_bounds__(256) void k_outproj(const bf16* __restrict__ Xb,
    const float* __restrict__ Wo, const float* __restrict__ bo,
    float* __restrict__ out) {
  __shared__ alignas(16) bf16 Ws2[128 * 32];
  const int t = threadIdx.x;
  const int mb = blockIdx.x >> 3;
  const int nb = blockIdx.x & 7;
  const int mbase = mb * 64, nbase = nb * 128;
  const int w = t >> 5, lane = t & 31, r = lane & 15, half = lane >> 4;
  const int mt = w & 1, nt = w >> 1;
  v8f acc[2][2];
#pragma unroll
  for (int a = 0; a < 2; ++a)
#pragma unroll
    for (int bcol = 0; bcol < 2; ++bcol) acc[a][bcol] = vzero8();
  for (int kb = 0; kb < DD; kb += 32) {
    {
      int row = t >> 1, c0 = (t & 1) * 16;
      const float4* g = (const float4*)(Wo + (size_t)(nbase + row) * DD + kb + c0);
      float4 f0 = g[0], f1 = g[1], f2 = g[2], f3 = g[3];
      bf16* d = &Ws2[row * 32 + c0];
      d[0]=(bf16)f0.x;  d[1]=(bf16)f0.y;  d[2]=(bf16)f0.z;  d[3]=(bf16)f0.w;
      d[4]=(bf16)f1.x;  d[5]=(bf16)f1.y;  d[6]=(bf16)f1.z;  d[7]=(bf16)f1.w;
      d[8]=(bf16)f2.x;  d[9]=(bf16)f2.y;  d[10]=(bf16)f2.z; d[11]=(bf16)f2.w;
      d[12]=(bf16)f3.x; d[13]=(bf16)f3.y; d[14]=(bf16)f3.z; d[15]=(bf16)f3.w;
    }
    __syncthreads();
    v16bf a0 = ld_a(Xb + (size_t)(mbase + mt * 32) * DD + kb, DD, half, r);
    v16bf a1 = ld_a(Xb + (size_t)(mbase + mt * 32 + 16) * DD + kb, DD, half, r);
    v16bf b0 = ld_b(&Ws2[(nt * 32 + 0) * 32], 32, half, r);
    v16bf b1 = ld_b(&Ws2[(nt * 32 + 16) * 32], 32, half, r);
    acc[0][0] = wmma_bf16(a0, b0, acc[0][0]);
    acc[0][1] = wmma_bf16(a0, b1, acc[0][1]);
    acc[1][0] = wmma_bf16(a1, b0, acc[1][0]);
    acc[1][1] = wmma_bf16(a1, b1, acc[1][1]);
    __syncthreads();
  }
#pragma unroll
  for (int ms = 0; ms < 2; ++ms)
#pragma unroll
    for (int ns = 0; ns < 2; ++ns) {
      int ng = nbase + nt * 32 + ns * 16 + r;
      float bb = bo[ng];
#pragma unroll
      for (int v = 0; v < 8; ++v) {
        int mg = mbase + mt * 32 + ms * 16 + v + 8 * half;
        out[(size_t)mg * DD + ng] = acc[ms][ns][v] + bb;
      }
    }
}

// ---------------- entropy -> certainty ; P0 = (avg+eps)^(1/TEMP) ----------------
__global__ __launch_bounds__(256) void k_entropy_p0(const float* __restrict__ avg,
    const float* __restrict__ cert, float* __restrict__ P0,
    float* __restrict__ cert_out) {
  __shared__ float red[256];
  const int row = blockIdx.x;           // B*N
  const int t = threadIdx.x;
  const size_t base = (size_t)row * NN;
  float hs = 0.0f;
  for (int j = t; j < NN; j += 256) {
    float a = avg[base + j];
    float ps = fmaxf(a, EPSF);
    hs -= ps * __logf(ps);
    P0[base + j] = __powf(a + EPSF, 10.0f);   // exp(log(a+eps)/0.1)
  }
  red[t] = hs;
  __syncthreads();
  for (int s = 128; s > 0; s >>= 1) {
    if (t < s) red[t] += red[t + s];
    __syncthreads();
  }
  if (t == 0) {
    float Hent = red[0];
    float upd = 1.0f / (1.0f + __expf(-(__logf((float)NN) - Hent)));
    cert_out[row] = fmaxf(cert[row], upd);
  }
}

// ---------------- fills ----------------
__global__ void k_fill4(float4* __restrict__ p, int n4, float val) {
  int i = blockIdx.x * 256 + threadIdx.x;
  if (i < n4) p[i] = make_float4(val, val, val, val);
}

// ---------------- Sinkhorn: row phase r_i = r_i / (r_i * (P0 c)_i + eps) ----------------
__global__ __launch_bounds__(256) void k_sink_row(const float* __restrict__ P0,
    const float* __restrict__ c, float* __restrict__ rr) {
  __shared__ float red[256];
  const int i = blockIdx.x;             // B*N
  const int t = threadIdx.x;
  const int b = i >> 11;
  const size_t base = (size_t)i * NN;
  const float* cb = c + (size_t)b * NN;
  float acc = 0.0f;
  for (int j = t; j < NN; j += 256) acc += P0[base + j] * cb[j];
  red[t] = acc;
  __syncthreads();
  for (int s = 128; s > 0; s >>= 1) {
    if (t < s) red[t] += red[t + s];
    __syncthreads();
  }
  if (t == 0) {
    float rv = rr[i];
    rr[i] = rv / (rv * red[0] + EPSF);
  }
}

// ---------------- Sinkhorn: partial column sums T_j += sum_i r_i P0_ij ----------------
__global__ __launch_bounds__(256) void k_sink_colsum(const float* __restrict__ P0,
    const float* __restrict__ rr, float* __restrict__ T) {
  const int cg = blockIdx.x;            // b*8 + colgroup (16)
  const int ys = blockIdx.y;            // 16 row splits
  const int b = cg >> 3;
  const int j = ((cg & 7) << 8) + threadIdx.x;
  const float* rb = rr + (size_t)b * NN;
  const size_t pb = (size_t)b * NN * NN;
  float acc = 0.0f;
  const int i0 = ys * 128;
  for (int i = i0; i < i0 + 128; ++i) acc += rb[i] * P0[pb + (size_t)i * NN + j];
  atomicAdd(&T[(size_t)b * NN + j], acc);
}

__global__ void k_sink_cupdate(const float* __restrict__ T, float* __restrict__ c) {
  int i = blockIdx.x * 256 + threadIdx.x;
  if (i < BB * NN) {
    float cc = c[i];
    c[i] = cc / (cc * T[i] + EPSF);
  }
}

// ---------------- argmax_j P0_ij * c_j (first occurrence) ----------------
__global__ __launch_bounds__(256) void k_argmax(const float* __restrict__ P0,
    const float* __restrict__ c, int* __restrict__ np) {
  __shared__ float bv[256];
  __shared__ int bi[256];
  const int i = blockIdx.x;
  const int t = threadIdx.x;
  const int b = i >> 11;
  const size_t base = (size_t)i * NN;
  const float* cb = c + (size_t)b * NN;
  float best = -1.0f;
  int bidx = NN;
  for (int j = t; j < NN; j += 256) {
    float v = P0[base + j] * cb[j];
    if (v > best) { best = v; bidx = j; }
  }
  bv[t] = best;
  bi[t] = bidx;
  __syncthreads();
  for (int s = 128; s > 0; s >>= 1) {
    if (t < s) {
      if (bv[t + s] > bv[t] || (bv[t + s] == bv[t] && bi[t + s] < bi[t])) {
        bv[t] = bv[t + s];
        bi[t] = bi[t + s];
      }
    }
    __syncthreads();
  }
  if (t == 0) np[i] = bi[0];
}

__global__ void k_compose(const long long* __restrict__ perm,
                          const int* __restrict__ np, long long* __restrict__ comp) {
  int i = blockIdx.x * 256 + threadIdx.x;
  if (i < BB * NN) {
    int b = i >> 11;
    long long p = perm[i];
    comp[i] = (long long)np[b * NN + (int)p];
  }
}

extern "C" void kernel_launch(void* const* d_in, const int* in_sizes, int n_in,
                              void* d_out, int out_size, void* d_ws, size_t ws_size,
                              hipStream_t stream) {
  (void)in_sizes; (void)n_in; (void)out_size; (void)ws_size;
  const float* data = (const float*)d_in[0];
  const float* cert = (const float*)d_in[1];
  const long long* perm = (const long long*)d_in[2];
  const float* Wq = (const float*)d_in[3];
  const float* bq = (const float*)d_in[4];
  const float* Wk = (const float*)d_in[5];
  const float* bk = (const float*)d_in[6];
  const float* Wv = (const float*)d_in[7];
  const float* bv = (const float*)d_in[8];
  const float* Wo = (const float*)d_in[9];
  const float* bo = (const float*)d_in[10];

  char* ws = (char*)d_ws;
  size_t off = 0;
  auto walloc = [&](size_t bytes) -> void* {
    void* p = ws + off;
    off += (bytes + 255) & ~(size_t)255;
    return p;
  };
  const size_t nQKV = (size_t)BB * HH * NN * HDIM;           // 4,194,304 elems
  bf16*  Q       = (bf16*)walloc(nQKV * 2);
  bf16*  K       = (bf16*)walloc(nQKV * 2);
  bf16*  V       = (bf16*)walloc(nQKV * 2);
  bf16*  Vt      = (bf16*)walloc(nQKV * 2);
  bf16*  attnout = (bf16*)walloc((size_t)BB * NN * DD * 2);
  float* Mst     = (float*)walloc((size_t)BB * HH * NN * 4);
  float* Lst     = (float*)walloc((size_t)BB * HH * NN * 4);
  float* avg     = (float*)walloc((size_t)BB * NN * NN * 4);  // 32 MB
  float* P0      = (float*)walloc((size_t)BB * NN * NN * 4);  // 32 MB
  float* rvec    = (float*)walloc((size_t)BB * NN * 4);
  float* cvec    = (float*)walloc((size_t)BB * NN * 4);
  float* Tvec    = (float*)walloc((size_t)BB * NN * 4);
  int*   newperm = (int*)walloc((size_t)BB * NN * 4);

  float* out_main = (float*)d_out;
  float* cert_out = out_main + (size_t)BB * NN * DD;
  long long* comp = (long long*)(out_main + (size_t)BB * NN * DD + (size_t)BB * NN);

  // zero avg_attn accumulator (32 MB -> 2,097,152 float4)
  k_fill4<<<8192, 256, 0, stream>>>((float4*)avg, 2097152, 0.0f);
  // QKV projection + V transpose
  k_qkv<<<512, 256, 0, stream>>>(data, Wq, bq, Wk, bk, Wv, bv, Q, K, V);
  k_transV<<<1024, 256, 0, stream>>>(V, Vt);
  // attention (two sweeps)
  k_attn_stats<<<512, 256, 0, stream>>>(Q, K, Mst, Lst);
  k_attn_apply<<<512, 256, 0, stream>>>(Q, K, Vt, Mst, Lst, avg, attnout);
  // output projection
  k_outproj<<<512, 256, 0, stream>>>(attnout, Wo, bo, out_main);
  // entropy/certainty + P0
  k_entropy_p0<<<BB * NN, 256, 0, stream>>>(avg, cert, P0, cert_out);
  // Sinkhorn via diagonal scale vectors (P0 stays L2-resident)
  k_fill4<<<4, 256, 0, stream>>>((float4*)rvec, (BB * NN) / 4, 1.0f);
  k_fill4<<<4, 256, 0, stream>>>((float4*)cvec, (BB * NN) / 4, 1.0f);
  for (int it = 0; it < 20; ++it) {
    k_sink_row<<<BB * NN, 256, 0, stream>>>(P0, cvec, rvec);
    k_fill4<<<4, 256, 0, stream>>>((float4*)Tvec, (BB * NN) / 4, 0.0f);
    k_sink_colsum<<<dim3(16, 16), 256, 0, stream>>>(P0, rvec, Tvec);
    k_sink_cupdate<<<16, 256, 0, stream>>>(Tvec, cvec);
  }
  // argmax + permutation composition
  k_argmax<<<BB * NN, 256, 0, stream>>>(P0, cvec, newperm);
  k_compose<<<16, 256, 0, stream>>>(perm, newperm, comp);
}
